// NeuralODE_57071525429649
// MI455X (gfx1250) — matrix operand order repaired and checked by reading, not compiled, not run on hardware
//
#include <hip/hip_runtime.h>
#include <hip/hip_bf16.h>
#include <cmath>

// ---------------------------------------------------------------------------
// Neural ODE (Dopri5, 64 steps) on MI455X / gfx1250.
//   f(t,y) = -(W2 @ tanh(W1 @ y + (1-t)*wt + b1) + b2)
// Strategy: fp16 weights (64 MB total -> resident in 192 MB L2), batch-1 GEMV
// expressed as V_WMMA_F32_16X16X32_F16 with the input vector broadcast across
// the M dimension and 16 weight rows in the B operand. fp32 accumulation.
// ---------------------------------------------------------------------------

typedef __attribute__((ext_vector_type(16))) _Float16 v16h;
typedef __attribute__((ext_vector_type(8)))  _Float16 v8h;
typedef __attribute__((ext_vector_type(4)))  _Float16 v4h;
typedef __attribute__((ext_vector_type(8)))  float    v8f;

#define D_DIM   2048
#define H_DIM   8192
#define N_STEPS 64

// ------------------------------ fp32 -> fp16 -------------------------------
__global__ __launch_bounds__(256)
void cvt_f32_f16(const float* __restrict__ s, _Float16* __restrict__ d, int n) {
  int i = (blockIdx.x * 256 + threadIdx.x) * 4;
  if (i < n) {
    float4 v = *(const float4*)(s + i);
    v4h o;
    o[0] = (_Float16)v.x; o[1] = (_Float16)v.y;
    o[2] = (_Float16)v.z; o[3] = (_Float16)v.w;
    *(v4h*)(d + i) = o;
  }
}

// -------- wave-level GEMV tile: 16 output rows, K-contraction via WMMA -----
// yh:  K-length fp16 vector in LDS (broadcast into A, all 16 M rows equal)
// wrow: base of this lane's weight row (row-major, fp16). B column n = row n.
// Returns dot(W[row], yh) in fp32 for row = nbase + (lane & 15).
__device__ inline float wmma_gemv_tile(const _Float16* __restrict__ wrow,
                                       const _Float16* __restrict__ yh,
                                       int K, int hi) {
  v8f acc = {};
  #pragma unroll 4
  for (int kc = 0; kc < K; kc += 32) {
    // ---- A operand (16x32 f16): broadcast vector.
    // lanes 0-15: halves[0..15]=y[kc+0..15]
    // lanes16-31: halves[0..15]=y[kc+16..31]
    v8h alo = *(const v8h*)(yh + kc + hi * 16);
    v8h ahi = *(const v8h*)(yh + kc + hi * 16 + 8);
    v16h a;
    #pragma unroll
    for (int j = 0; j < 8; ++j) { a[j] = alo[j]; a[8 + j] = ahi[j]; }
    // ---- B operand (32x16 f16): column (lane&15) = weight row, contiguous K.
    // lanes 0-15 hold K=kc..kc+15, lanes 16-31 hold K=kc+16..kc+31.
    const _Float16* bp = wrow + kc + hi * 16;
    v8h blo = *(const v8h*)(bp);
    v8h bhi = *(const v8h*)(bp + 8);
    v16h b;
    #pragma unroll
    for (int j = 0; j < 8; ++j) { b[j] = blo[j]; b[8 + j] = bhi[j]; }
    acc = __builtin_amdgcn_wmma_f32_16x16x32_f16(
        /*neg_a=*/false, a, /*neg_b=*/false, b,
        /*c_mod=*/(short)0, acc, /*reuse_a=*/false, /*reuse_b=*/false);
  }
  // All M rows identical; C VGPR0: lanes 0-15 -> (M=0,N=lane),
  // lanes 16-31 -> (M=8,N=lane-16). Either way acc[0] = out[nbase+(lane&15)].
  return acc[0];
}

// ---------------- stage kernel 1: yi = y + dt*sum(a_j k_j);  --------------
// ---------------- h = tanh(W1 @ yi + s*wt + b1)  (h stored fp16) ----------
__global__ __launch_bounds__(256)
void gemv1_tanh(const _Float16* __restrict__ W1h,
                const float* __restrict__ y,
                const float* __restrict__ k0, const float* __restrict__ k1,
                const float* __restrict__ k2, const float* __restrict__ k3,
                const float* __restrict__ k4,
                float a0, float a1, float a2, float a3, float a4, int nk,
                float s_time,
                const float* __restrict__ wt, const float* __restrict__ bias1,
                _Float16* __restrict__ h_out) {
  __shared__ _Float16 yh[D_DIM];
  int tid = threadIdx.x;
  // Fused RK stage-input combination, redundantly per block (trivial cost).
  for (int r = tid; r < D_DIM; r += 256) {
    float v = y[r];
    if (nk > 0) v += a0 * k0[r];
    if (nk > 1) v += a1 * k1[r];
    if (nk > 2) v += a2 * k2[r];
    if (nk > 3) v += a3 * k3[r];
    if (nk > 4) v += a4 * k4[r];
    yh[r] = (_Float16)v;
  }
  __syncthreads();

  int wave = tid >> 5;
  int lane = tid & 31;
  int hi   = lane >> 4;
  int tile = blockIdx.x * 8 + wave;        // 512 tiles cover H=8192 rows
  int row  = tile * 16 + (lane & 15);
  const _Float16* wrow = W1h + (size_t)row * D_DIM;

  float dot = wmma_gemv_tile(wrow, yh, D_DIM, hi);

  if (lane < 16) {
    float pre = dot + s_time * wt[row] + bias1[row];
    h_out[row] = (_Float16)tanhf(pre);
  }
}

// ---------------- stage kernel 2: k_i = -(W2 @ h + b2) --------------------
// On the last stage (do_update) also: y += dt * sum(B_i k_i)  (in place).
__global__ __launch_bounds__(256)
void gemv2_stage(const _Float16* __restrict__ W2h,
                 const _Float16* __restrict__ hvec,
                 const float* __restrict__ bias2,
                 float* __restrict__ kout,
                 float* __restrict__ y,
                 const float* __restrict__ k0, const float* __restrict__ k2,
                 const float* __restrict__ k3, const float* __restrict__ k4,
                 float c0, float c2, float c3, float c4, float c5,
                 int do_update) {
  __shared__ _Float16 hs[H_DIM];
  int tid = threadIdx.x;
  #pragma unroll
  for (int r = tid * 4; r < H_DIM; r += 256 * 4)
    *(v4h*)(hs + r) = *(const v4h*)(hvec + r);
  __syncthreads();

  int wave = tid >> 5;
  int lane = tid & 31;
  int hi   = lane >> 4;
  int tile = blockIdx.x * 8 + wave;        // 128 tiles cover D=2048 rows
  int row  = tile * 16 + (lane & 15);
  const _Float16* wrow = W2h + (size_t)row * H_DIM;

  float dot = wmma_gemv_tile(wrow, hs, H_DIM, hi);

  if (lane < 16) {
    float val = -(dot + bias2[row]);
    kout[row] = val;
    if (do_update) {
      // b[1] == 0 in the Dopri5 tableau, so k1 is skipped.
      y[row] = y[row] + c0 * k0[row] + c2 * k2[row]
                      + c3 * k3[row] + c4 * k4[row] + c5 * val;
    }
  }
}

// ---------------------------------------------------------------------------
extern "C" void kernel_launch(void* const* d_in, const int* in_sizes, int n_in,
                              void* d_out, int out_size, void* d_ws, size_t ws_size,
                              hipStream_t stream) {
  const float* x  = (const float*)d_in[0];   // [2048]
  const float* W1 = (const float*)d_in[1];   // [8192,2048]
  const float* wt = (const float*)d_in[2];   // [8192]
  const float* b1 = (const float*)d_in[3];   // [8192]
  const float* W2 = (const float*)d_in[4];   // [2048,8192]
  const float* b2 = (const float*)d_in[5];   // [2048]

  // Workspace layout (needs ~64 MB + 72 KB).
  char* p = (char*)d_ws;
  _Float16* W1h = (_Float16*)p; p += (size_t)H_DIM * D_DIM * sizeof(_Float16);
  _Float16* W2h = (_Float16*)p; p += (size_t)D_DIM * H_DIM * sizeof(_Float16);
  _Float16* hb  = (_Float16*)p; p += (size_t)H_DIM * sizeof(_Float16);
  float*    y   = (float*)p;    p += (size_t)D_DIM * sizeof(float);
  float*    k[6];
  for (int i = 0; i < 6; ++i) { k[i] = (float*)p; p += (size_t)D_DIM * sizeof(float); }

  const int nW = H_DIM * D_DIM;
  cvt_f32_f16<<<nW / 1024, 256, 0, stream>>>(W1, W1h, nW);
  cvt_f32_f16<<<nW / 1024, 256, 0, stream>>>(W2, W2h, nW);
  hipMemcpyAsync(y, x, D_DIM * sizeof(float), hipMemcpyDeviceToDevice, stream);

  const float dt = 1.0f / N_STEPS;
  static const float Cc[6] = {0.0f, 0.2f, 0.3f, 0.8f, 8.0f / 9.0f, 1.0f};
  static const double At[6][5] = {
      {0, 0, 0, 0, 0},
      {1.0 / 5, 0, 0, 0, 0},
      {3.0 / 40, 9.0 / 40, 0, 0, 0},
      {44.0 / 45, -56.0 / 15, 32.0 / 9, 0, 0},
      {19372.0 / 6561, -25360.0 / 2187, 64448.0 / 6561, -212.0 / 729, 0},
      {9017.0 / 3168, -355.0 / 33, 46732.0 / 5247, 49.0 / 176, -5103.0 / 18656}};
  static const double Bt[6] = {35.0 / 384, 0.0, 500.0 / 1113,
                               125.0 / 192, -2187.0 / 6784, 11.0 / 84};

  for (int st = 0; st < N_STEPS; ++st) {
    float t = (float)st * dt;
    for (int i = 0; i < 6; ++i) {
      float s = 1.0f - (t + Cc[i] * dt);   // reversed-time argument
      float a[5] = {0.f, 0.f, 0.f, 0.f, 0.f};
      for (int j = 0; j < i; ++j) a[j] = dt * (float)At[i][j];

      gemv1_tanh<<<64, 256, 0, stream>>>(
          W1h, y, k[0], k[1], k[2], k[3], k[4],
          a[0], a[1], a[2], a[3], a[4], i, s, wt, b1, hb);

      int last = (i == 5) ? 1 : 0;
      gemv2_stage<<<16, 256, 0, stream>>>(
          W2h, hb, b2, k[i], y, k[0], k[2], k[3], k[4],
          dt * (float)Bt[0], dt * (float)Bt[2], dt * (float)Bt[3],
          dt * (float)Bt[4], dt * (float)Bt[5], last);
    }
  }

  hipMemcpyAsync(d_out, y, D_DIM * sizeof(float), hipMemcpyDeviceToDevice, stream);
}
